// SelfAttention_47270410059804
// MI455X (gfx1250) — matrix operand order repaired and checked by reading, not compiled, not run on hardware
//
#include <hip/hip_runtime.h>
#include <hip/hip_bf16.h>
#include <math.h>

typedef __attribute__((ext_vector_type(16))) _Float16 v16h;
typedef __attribute__((ext_vector_type(8)))  _Float16 v8h;
typedef __attribute__((ext_vector_type(8)))  float    v8f;
typedef __attribute__((ext_vector_type(4)))  float    v4f;

#define L_DIM 2048
#define C_DIM 256
#define B_DIM 8

// ---------- fragment loaders matching CDNA5 WMMA 16-bit layouts ----------
// A 16x32 (MxK), source row-major MxK, stride in halves.
// lane<16: K[0..7] then K[16..23]; lane>=16: K[8..15] then K[24..31].
__device__ __forceinline__ v16h frag_a_f16(const _Float16* __restrict__ base, int stride, int lane) {
  const int r = lane & 15;
  const int koff = (lane >> 4) << 3;
  const _Float16* p = base + (size_t)r * stride + koff;
  v16h f;
  ((v8h*)&f)[0] = *(const v8h*)(p);
  ((v8h*)&f)[1] = *(const v8h*)(p + 16);
  return f;
}

// B 32x16 (KxN), source stored as row-major NxK (i.e. B^T), stride in halves.
// lane = column n = lane&15; lane<16: K[0..7] then K[16..23]; lane>=16: K[8..15] then K[24..31].
__device__ __forceinline__ v16h frag_b_f16(const _Float16* __restrict__ base, int stride, int lane) {
  const int n = lane & 15;
  const int kb = (lane >> 4) << 3;
  const _Float16* p = base + (size_t)n * stride + kb;
  v16h f;
  ((v8h*)&f)[0] = *(const v8h*)(p);
  ((v8h*)&f)[1] = *(const v8h*)(p + 16);
  return f;
}

// =========================================================================
// Kernel 0: bulk f32 -> f16 convert (one shot; removes cvt from GEMM loops).
// Each thread converts 8 contiguous elements.
// =========================================================================
__global__ void __launch_bounds__(256)
cvt_f32_to_f16_kernel(const float* __restrict__ src, _Float16* __restrict__ dst) {
  const size_t i = ((size_t)blockIdx.x * 256 + threadIdx.x) * 8;
  v4f a = *(const v4f*)(src + i);
  v4f b = *(const v4f*)(src + i + 4);
  v8h o;
#pragma unroll
  for (int e = 0; e < 4; ++e) {
    o[e]     = (_Float16)a[e];
    o[e + 4] = (_Float16)b[e];
  }
  *(v8h*)(dst + i) = o;
}

// =========================================================================
// Common GEMM body: 256 threads = 8 waves, block tile 128(rows) x 256(cols),
// wave tile 64x64 = 4x4 WMMA tiles (16 WMMAs per 16 b128 fragment loads).
// =========================================================================
#define GEMM_PROLOGUE()                                              \
  const int tid  = threadIdx.x;                                      \
  const int lane = tid & 31;                                         \
  const int wave = tid >> 5;                                         \
  const int rowBase = blockIdx.y * 128 + (wave & 1) * 64;            \
  const int colBase = blockIdx.x * 256 + (wave >> 1) * 64;           \
  v8f acc[4][4];                                                     \
  _Pragma("unroll")                                                  \
  for (int i = 0; i < 4; ++i)                                        \
    _Pragma("unroll")                                                \
    for (int j = 0; j < 4; ++j)                                      \
      _Pragma("unroll")                                              \
      for (int e = 0; e < 8; ++e) acc[i][j][e] = 0.0f;

#define GEMM_KLOOP(Abase, Astride, Bbase, Bstride, KLEN)             \
  for (int k = 0; k < (KLEN); k += 32) {                             \
    v16h a[4], b[4];                                                 \
    _Pragma("unroll")                                                \
    for (int i = 0; i < 4; ++i)                                      \
      a[i] = frag_a_f16((Abase) + (size_t)(rowBase + 16 * i) * (Astride) + k, (Astride), lane); \
    _Pragma("unroll")                                                \
    for (int j = 0; j < 4; ++j)                                      \
      b[j] = frag_b_f16((Bbase) + (size_t)(colBase + 16 * j) * (Bstride) + k, (Bstride), lane); \
    _Pragma("unroll")                                                \
    for (int i = 0; i < 4; ++i)                                      \
      _Pragma("unroll")                                              \
      for (int j = 0; j < 4; ++j)                                    \
        acc[i][j] = __builtin_amdgcn_wmma_f32_16x16x32_f16(          \
            false, a[i], false, b[j], (short)0, acc[i][j], false, false); \
  }

// =========================================================================
// Kernel 1: QKV projection (f16 GEMM). out[n,m] = sum_l xh[n,l]*Wh[m,l] + b[m]
// n = b*256+c over 2048 rows, m over 2048 cols, K = 2048.
// blockIdx.z selects q/k/v. q,k stored transposed as [b][m][c]; v as [b][c][m].
// =========================================================================
__global__ void __launch_bounds__(256)
qkv_proj_kernel(const _Float16* __restrict__ xh,
                const _Float16* __restrict__ Wqh, const float* __restrict__ bq,
                const _Float16* __restrict__ Wkh, const float* __restrict__ bk,
                const _Float16* __restrict__ Wvh, const float* __restrict__ bv,
                _Float16* __restrict__ qT, _Float16* __restrict__ kT,
                _Float16* __restrict__ vO) {
  const int which = blockIdx.z;
  const _Float16* W  = (which == 0) ? Wqh : (which == 1) ? Wkh : Wvh;
  const float* bias  = (which == 0) ? bq  : (which == 1) ? bk  : bv;

  GEMM_PROLOGUE()
  GEMM_KLOOP(xh, L_DIM, W, L_DIM, L_DIM)

  const int cN  = lane & 15;
  const int rHi = (lane >> 4) * 8;
#pragma unroll
  for (int i = 0; i < 4; ++i) {
#pragma unroll
    for (int j = 0; j < 4; ++j) {
      const int m = colBase + 16 * j + cN;
      const float bm = bias[m];
#pragma unroll
      for (int e = 0; e < 8; ++e) {
        const int n = rowBase + 16 * i + rHi + e;
        const float val = acc[i][j][e] + bm;
        const int bb = n >> 8;     // batch
        const int cc = n & 255;    // channel
        if (which < 2) {
          _Float16* dst = (which == 0) ? qT : kT;
          dst[(size_t)bb * (L_DIM * C_DIM) + (size_t)m * C_DIM + cc] = (_Float16)val;
        } else {
          vO[(size_t)bb * (C_DIM * L_DIM) + (size_t)cc * L_DIM + m] = (_Float16)val;
        }
      }
    }
  }
}

// =========================================================================
// Kernel 2: attention scores for one batch.
// S[l,m] = scale * sum_c qT[l,c] * kT[m,c]   (K = 256)
// =========================================================================
__global__ void __launch_bounds__(256)
scores_kernel(const _Float16* __restrict__ qT, const _Float16* __restrict__ kT,
              float* __restrict__ S, float scale) {
  GEMM_PROLOGUE()
  GEMM_KLOOP(qT, C_DIM, kT, C_DIM, C_DIM)

  const int cN  = lane & 15;
  const int rHi = (lane >> 4) * 8;
#pragma unroll
  for (int i = 0; i < 4; ++i)
#pragma unroll
    for (int j = 0; j < 4; ++j)
#pragma unroll
      for (int e = 0; e < 8; ++e) {
        const int l = rowBase + 16 * i + rHi + e;
        const int m = colBase + 16 * j + cN;
        S[(size_t)l * L_DIM + m] = acc[i][j][e] * scale;
      }
}

// =========================================================================
// Kernel 3: row softmax. One block (256 threads) per row of 2048 f32.
// Writes f16 attn row-major [L, M].
// =========================================================================
__global__ void __launch_bounds__(256)
softmax_kernel(const float* __restrict__ S, _Float16* __restrict__ attn) {
  const int row = blockIdx.x;
  const int tid = threadIdx.x;
  const float* s = S + (size_t)row * L_DIM;
  _Float16* a = attn + (size_t)row * L_DIM;
  __shared__ float red[256];

  float vals[8];
  float mx = -INFINITY;
#pragma unroll
  for (int i = 0; i < 8; ++i) {
    vals[i] = s[tid + 256 * i];
    mx = fmaxf(mx, vals[i]);
  }
  red[tid] = mx;
  __syncthreads();
  for (int off = 128; off > 0; off >>= 1) {
    if (tid < off) red[tid] = fmaxf(red[tid], red[tid + off]);
    __syncthreads();
  }
  mx = red[0];
  __syncthreads();

  float sum = 0.0f;
#pragma unroll
  for (int i = 0; i < 8; ++i) {
    vals[i] = __expf(vals[i] - mx);
    sum += vals[i];
  }
  red[tid] = sum;
  __syncthreads();
  for (int off = 128; off > 0; off >>= 1) {
    if (tid < off) red[tid] += red[tid + off];
    __syncthreads();
  }
  const float inv = 1.0f / red[0];
#pragma unroll
  for (int i = 0; i < 8; ++i)
    a[tid + 256 * i] = (_Float16)(vals[i] * inv);
}

// =========================================================================
// Kernel 4: out[c,l] = sum_m v[c,m] * attn[l,m]   (K = 2048), f32 output.
// rows = C (256), cols = L (2048).
// =========================================================================
__global__ void __launch_bounds__(256)
out_gemm_kernel(const _Float16* __restrict__ vM, const _Float16* __restrict__ attn,
                float* __restrict__ out) {
  GEMM_PROLOGUE()
  GEMM_KLOOP(vM, L_DIM, attn, L_DIM, L_DIM)

  const int cN  = lane & 15;
  const int rHi = (lane >> 4) * 8;
#pragma unroll
  for (int i = 0; i < 4; ++i)
#pragma unroll
    for (int j = 0; j < 4; ++j)
#pragma unroll
      for (int e = 0; e < 8; ++e) {
        const int c = rowBase + 16 * i + rHi + e;
        const int l = colBase + 16 * j + cN;
        out[(size_t)c * L_DIM + l] = acc[i][j][e];
      }
}

// =========================================================================
extern "C" void kernel_launch(void* const* d_in, const int* in_sizes, int n_in,
                              void* d_out, int out_size, void* d_ws, size_t ws_size,
                              hipStream_t stream) {
  const float* x  = (const float*)d_in[0];
  const float* Wq = (const float*)d_in[1];
  const float* bq = (const float*)d_in[2];
  const float* Wk = (const float*)d_in[3];
  const float* bk = (const float*)d_in[4];
  const float* Wv = (const float*)d_in[5];
  const float* bv = (const float*)d_in[6];
  float* out = (float*)d_out;

  char* ws = (char*)d_ws;
  size_t off = 0;
  auto wsAlloc = [&](size_t bytes) -> char* {
    char* p = ws + off;
    off = (off + bytes + 255) & ~(size_t)255;
    return p;
  };

  const size_t matElems = (size_t)L_DIM * L_DIM;        // 4,194,304 (x, W, per-batch S/attn)
  const size_t qkvElems = (size_t)B_DIM * L_DIM * C_DIM; // 4,194,304

  _Float16* xh   = (_Float16*)wsAlloc(matElems * sizeof(_Float16));   // [N][L]
  _Float16* Wqh  = (_Float16*)wsAlloc(matElems * sizeof(_Float16));   // [M][L]
  _Float16* Wkh  = (_Float16*)wsAlloc(matElems * sizeof(_Float16));
  _Float16* Wvh  = (_Float16*)wsAlloc(matElems * sizeof(_Float16));
  _Float16* qT   = (_Float16*)wsAlloc(qkvElems * sizeof(_Float16));   // [B][L][C]
  _Float16* kT   = (_Float16*)wsAlloc(qkvElems * sizeof(_Float16));   // [B][M][C]
  _Float16* vbuf = (_Float16*)wsAlloc(qkvElems * sizeof(_Float16));   // [B][C][M]
  float*    Sbuf = (float*)wsAlloc(matElems * sizeof(float));         // per-batch, reused
  _Float16* attn = (_Float16*)wsAlloc(matElems * sizeof(_Float16));   // per-batch, reused

  const dim3 blk(256);
  const int cvtBlocks = (int)(matElems / (256 * 8));  // 2048
  const float scale = 1.0f / sqrtf((float)L_DIM);

  cvt_f32_to_f16_kernel<<<dim3(cvtBlocks), blk, 0, stream>>>(x,  xh);
  cvt_f32_to_f16_kernel<<<dim3(cvtBlocks), blk, 0, stream>>>(Wq, Wqh);
  cvt_f32_to_f16_kernel<<<dim3(cvtBlocks), blk, 0, stream>>>(Wk, Wkh);
  cvt_f32_to_f16_kernel<<<dim3(cvtBlocks), blk, 0, stream>>>(Wv, Wvh);

  // block tile 128 rows x 256 cols
  qkv_proj_kernel<<<dim3(8, 16, 3), blk, 0, stream>>>(xh, Wqh, bq, Wkh, bk, Wvh, bv,
                                                      qT, kT, vbuf);

  for (int b = 0; b < B_DIM; ++b) {
    const size_t mOff = (size_t)b * L_DIM * C_DIM;
    scores_kernel<<<dim3(8, 16), blk, 0, stream>>>(qT + mOff, kT + mOff, Sbuf, scale);
    softmax_kernel<<<dim3(L_DIM), blk, 0, stream>>>(Sbuf, attn);
    out_gemm_kernel<<<dim3(8, 2), blk, 0, stream>>>(vbuf + (size_t)b * C_DIM * L_DIM, attn,
                                                    out + (size_t)b * C_DIM * L_DIM);
  }
}
